// LinCos_41686952575613
// MI455X (gfx1250) — matrix operand order repaired and checked by reading, not compiled, not run on hardware
//
#include <hip/hip_runtime.h>

typedef __attribute__((ext_vector_type(16))) __bf16 v16bf;
typedef __attribute__((ext_vector_type(8)))  float  v8f;
typedef __attribute__((ext_vector_type(4)))  float  f32x4;

#define B_ROWS   65536
#define IN_DIM   512
#define OUT_DIM  2048
#define BM 128
#define BN 128
#define BK 32
#define LDS_STRIDE 40            // 32 bf16 + 16-byte pad
#define KSTEPS (IN_DIM / BK)     // 16

__launch_bounds__(256)
__global__ void lincos_wmma(const float* __restrict__ x,
                            const float* __restrict__ w,
                            const float* __restrict__ bias,
                            float* __restrict__ out)
{
    __shared__ __align__(16) __bf16 As[BM * LDS_STRIDE];
    __shared__ __align__(16) __bf16 Bs[BN * LDS_STRIDE];
    __shared__ float axs[BM];   // 1/max(||x_row||, eps)
    __shared__ float bws[BN];   // 1/max(||w_row||, eps)

    const int tid   = threadIdx.x;
    const int lane  = tid & 31;
    const int wav   = tid >> 5;       // 0..7
    const int wr    = wav >> 2;       // 0..1 : 64-row group
    const int wc    = wav & 3;        // 0..3 : 32-col group
    const int lmod  = lane & 15;
    const int khalf = lane >> 4;      // 0..1

    const int rowBase = blockIdx.y * BM;
    const int colBase = blockIdx.x * BN;

    // cooperative load mapping: 2 threads per row, 16 contiguous floats each
    const int ldRow  = tid >> 1;          // 0..127
    const int ldHalf = (tid & 1) * 16;    // 0 or 16 (within BK=32)

    const float* gx = x + (size_t)(rowBase + ldRow) * IN_DIM + ldHalf;
    const float* gw = w + (size_t)(colBase + ldRow) * IN_DIM + ldHalf;

    v8f acc[4][2];
    #pragma unroll
    for (int mi = 0; mi < 4; ++mi)
        #pragma unroll
        for (int ni = 0; ni < 2; ++ni)
            #pragma unroll
            for (int e = 0; e < 8; ++e)
                acc[mi][ni][e] = 0.0f;

    float sx = 0.0f, sw = 0.0f;

    // software-pipeline: prime register staging with k-slab 0
    f32x4 rx[4], rw[4];
    #pragma unroll
    for (int i = 0; i < 4; ++i) {
        rx[i] = *(const f32x4*)(gx + i * 4);
        rw[i] = *(const f32x4*)(gw + i * 4);
    }

    for (int kk = 0; kk < KSTEPS; ++kk) {
        // convert fp32->bf16, accumulate row sum-of-squares, store to LDS
        union { __bf16 h[16]; f32x4 q[2]; } px, pw;
        #pragma unroll
        for (int i = 0; i < 4; ++i) {
            #pragma unroll
            for (int j = 0; j < 4; ++j) {
                float fx = rx[i][j];
                float fw = rw[i][j];
                sx += fx * fx;
                sw += fw * fw;
                px.h[i * 4 + j] = (__bf16)fx;
                pw.h[i * 4 + j] = (__bf16)fw;
            }
        }
        *(f32x4*)&As[ldRow * LDS_STRIDE + ldHalf]     = px.q[0];
        *(f32x4*)&As[ldRow * LDS_STRIDE + ldHalf + 8] = px.q[1];
        *(f32x4*)&Bs[ldRow * LDS_STRIDE + ldHalf]     = pw.q[0];
        *(f32x4*)&Bs[ldRow * LDS_STRIDE + ldHalf + 8] = pw.q[1];
        __syncthreads();

        // prefetch next k-slab from global while WMMAs run on LDS
        if (kk + 1 < KSTEPS) {
            #pragma unroll
            for (int i = 0; i < 4; ++i) {
                rx[i] = *(const f32x4*)(gx + (kk + 1) * BK + i * 4);
                rw[i] = *(const f32x4*)(gw + (kk + 1) * BK + i * 4);
            }
        }

        // B fragments: lane n holds column n; K 0..15 (lanes 0-15) / 16..31 (16-31)
        union { f32x4 q[2]; v16bf v; } bf[2];
        #pragma unroll
        for (int ni = 0; ni < 2; ++ni) {
            const int rW = wc * 32 + ni * 16 + lmod;
            bf[ni].q[0] = *(const f32x4*)&Bs[rW * LDS_STRIDE + khalf * 16];
            bf[ni].q[1] = *(const f32x4*)&Bs[rW * LDS_STRIDE + khalf * 16 + 8];
        }

        // A fragments: lane m holds row m: K (khalf*16..+15)
        #pragma unroll
        for (int mi = 0; mi < 4; ++mi) {
            union { f32x4 q[2]; v16bf v; } af;
            const int rA = wr * 64 + mi * 16 + lmod;
            af.q[0] = *(const f32x4*)&As[rA * LDS_STRIDE + khalf * 16];
            af.q[1] = *(const f32x4*)&As[rA * LDS_STRIDE + khalf * 16 + 8];
            #pragma unroll
            for (int ni = 0; ni < 2; ++ni) {
                acc[mi][ni] = __builtin_amdgcn_wmma_f32_16x16x32_bf16(
                    false, af.v, false, bf[ni].v,
                    (short)0, acc[mi][ni], false, false);
            }
        }
        __syncthreads();
    }

    // finish norms: combine the two half-row partials (adjacent lanes)
    sx += __shfl_xor(sx, 1, 32);
    sw += __shfl_xor(sw, 1, 32);
    if ((tid & 1) == 0) {
        axs[ldRow] = 1.0f / fmaxf(sqrtf(sx), 1e-8f);
        bws[ldRow] = 1.0f / fmaxf(sqrtf(sw), 1e-8f);
    }
    __syncthreads();

    // epilogue: C layout — VGPR j: lanes 0-15 -> M=j, lanes 16-31 -> M=j+8; N=lane&15
    const float scale = 22.62741699796952f;   // sqrt(512)
    #pragma unroll
    for (int ni = 0; ni < 2; ++ni) {
        const int nl = wc * 32 + ni * 16 + lmod;
        const float bn = bias[colBase + nl];
        const float wn = bws[nl] * scale;
        #pragma unroll
        for (int mi = 0; mi < 4; ++mi) {
            #pragma unroll
            for (int j = 0; j < 8; ++j) {
                const int ml = wr * 64 + mi * 16 + j + khalf * 8;
                const float v = acc[mi][ni][j] * (axs[ml] * wn) + bn;
                out[(size_t)(rowBase + ml) * OUT_DIM + (colBase + nl)] = v;
            }
        }
    }
}

extern "C" void kernel_launch(void* const* d_in, const int* in_sizes, int n_in,
                              void* d_out, int out_size, void* d_ws, size_t ws_size,
                              hipStream_t stream)
{
    const float* x    = (const float*)d_in[0];
    const float* wgt  = (const float*)d_in[1];
    const float* bias = (const float*)d_in[2];
    float* out        = (float*)d_out;

    dim3 grid(OUT_DIM / BN, B_ROWS / BM);   // (16, 512): column-tile fastest for L2 reuse of x
    lincos_wmma<<<grid, 256, 0, stream>>>(x, wgt, bias, out);
}
